// GATEncoder_29996051595953
// MI455X (gfx1250) — matrix operand order repaired and checked by reading, not compiled, not run on hardware
//
#include <hip/hip_runtime.h>

// ---------------------------------------------------------------------------
// GATv2 encoder for gfx1250 (MI455X):
//   - WMMA f16/f32-acc GEMMs, 16x64 register-blocked per wave, B panel in LDS
//   - L2-resident edge phase, float4 gathers, folded segment-softmax
// ---------------------------------------------------------------------------

typedef _Float16 half8 __attribute__((ext_vector_type(8)));
typedef _Float16 v16h  __attribute__((ext_vector_type(16)));
typedef float    v8f   __attribute__((ext_vector_type(8)));

constexpr int N_   = 50000;
constexpr int E_   = 400000;
constexpr int F_   = 64;
constexpr int H_   = 3;
constexpr int C1_  = 64;
constexpr int HC1  = H_ * C1_;    // 192
constexpr int F2_  = F_ + HC1;    // 256
constexpr int C2_  = 128;
constexpr int HC2  = H_ * C2_;    // 384
constexpr int OUTC = F_ + C2_;    // 192
constexpr float SLOPE = 0.15f;

// order-preserving float <-> uint encoding for atomic segment-max
__device__ __forceinline__ unsigned fenc(float f) {
    unsigned u = __float_as_uint(f);
    return (u >> 31) ? ~u : (u | 0x80000000u);
}
__device__ __forceinline__ float fdec(unsigned u) {
    return (u >> 31) ? __uint_as_float(u & 0x7FFFFFFFu) : __uint_as_float(~u);
}
constexpr unsigned ENC_NEG_INF = 0x007FFFFFu;   // fenc(-inf)

__device__ __forceinline__ float lrelu(float v) { return (v > 0.f) ? v : v * SLOPE; }

// ---------------------------------------------------------------------------
__global__ void k_fill_u32(unsigned* __restrict__ p, unsigned v, long n) {
    long t = (long)blockIdx.x * blockDim.x + threadIdx.x;
    if (t < n) p[t] = v;
}

__global__ void k_cvt_f16(const float* __restrict__ s, _Float16* __restrict__ d, long n) {
    long t = (long)blockIdx.x * blockDim.x + threadIdx.x;
    if (t < n) d[t] = (_Float16)s[t];
}

// W (K x Nout, row-major) -> WT (Nout x K, f16) so B fragments load contiguously
__global__ void k_cvt_w_t(const float* __restrict__ W, _Float16* __restrict__ WT,
                          int K, int Nout) {
    long t = (long)blockIdx.x * blockDim.x + threadIdx.x;
    if (t >= (long)K * Nout) return;
    int n = (int)(t / K), k = (int)(t % K);
    WT[t] = (_Float16)W[(size_t)k * Nout + n];
}

// ---------------------------------------------------------------------------
// C[M x Nout] = A16[M x K] @ B16[K x Nout] + bias  (BT given as Nout x K)
// Block: 160 threads = 5 waves; each wave owns one 16-row M tile and computes
// a 16x64 strip (4 WMMA accumulators sharing one A fragment per k-step).
// The 64-column B panel (64 x K f16, <=32KB) is staged in LDS once per block
// and shared by all 5 waves.  grid = (M/80, Nout/64).
// ---------------------------------------------------------------------------
template <int K>
__global__ void k_gemm_wmma(const _Float16* __restrict__ A,
                            const _Float16* __restrict__ BT,
                            const float* __restrict__ bias,
                            float* __restrict__ Cm, int Nout) {
    __shared__ _Float16 lb[64 * K];           // B panel: rows nbase..nbase+63, K deep

    const int tid   = threadIdx.x;
    const int lane  = tid & 31;
    const int wave  = tid >> 5;               // 0..4
    const int nbase = blockIdx.y * 64;

    // cooperative stage of B panel into LDS (16B chunks)
    {
        const half8* gsrc = (const half8*)(BT + (size_t)nbase * K);
        half8*       ldst = (half8*)lb;
        const int total8 = 64 * K / 8;
        for (int i = tid; i < total8; i += 160) ldst[i] = gsrc[i];
    }
    __syncthreads();

    const int hs = lane >> 4;                 // half-wave select
    const int lm = lane & 15;

    // A fragment (16-bit 16x32): lanes 0-15 M=lm hold K 0..7 & 16..23;
    //                            lanes 16-31 hold K 8..15 & 24..31
    const int mrow = (blockIdx.x * 5 + wave) * 16 + lm;
    const _Float16* a0 = A + (size_t)mrow * K + hs * 8;

    // B fragments from LDS: lanes 0-15 N=lm hold K 0..7 & 16..23; lanes 16-31 K 8..15 & 24..31
    const _Float16* b0 = lb + (size_t)lm * K + hs * 8;

    v8f acc[4];
#pragma unroll
    for (int j = 0; j < 4; ++j) acc[j] = v8f{0.f,0.f,0.f,0.f,0.f,0.f,0.f,0.f};

#pragma unroll
    for (int k0 = 0; k0 < K; k0 += 32) {
        union { v16h v; half8 h[2]; } ua;
        ua.h[0] = *(const half8*)(a0 + k0);
        ua.h[1] = *(const half8*)(a0 + k0 + 16);
#pragma unroll
        for (int j = 0; j < 4; ++j) {
            union { v16h v; half8 h[2]; } ub;
            const _Float16* bj = b0 + (size_t)(j * 16) * K;
            ub.h[0] = *(const half8*)(bj + k0);
            ub.h[1] = *(const half8*)(bj + k0 + 16);
            acc[j] = __builtin_amdgcn_wmma_f32_16x16x32_f16(
                false, ua.v, false, ub.v, (short)0, acc[j], false, false);
        }
    }

    // C/D layout: VGPR r -> M = r + 8*hs, N = lm
    const int mtop = (blockIdx.x * 5 + wave) * 16;
#pragma unroll
    for (int j = 0; j < 4; ++j) {
        const int col = nbase + j * 16 + lm;
        const float bv = bias[col];
#pragma unroll
        for (int r = 0; r < 8; ++r) {
            int row = mtop + hs * 8 + r;
            Cm[(size_t)row * Nout + col] = acc[j][r] + bv;
        }
    }
}

// ---------------------------------------------------------------------------
// Edge pass 1: logits[e,h] = att_h . leaky_relu(xl[src] + xr[dst]); segment max
// ---------------------------------------------------------------------------
__global__ void k_edge_logits(const float* __restrict__ xl, const float* __restrict__ xr,
                              const float* __restrict__ att,
                              const int* __restrict__ src, const int* __restrict__ dst,
                              float* __restrict__ logits, unsigned* __restrict__ maxb,
                              int C) {
    long t = (long)blockIdx.x * blockDim.x + threadIdx.x;
    if (t >= (long)E_ * H_) return;
    int e = (int)(t / H_), h = (int)(t - (long)e * H_);
    int s = src[e], d = dst[e];
    const float4* pl = (const float4*)(xl + (size_t)s * H_ * C + (size_t)h * C);
    const float4* pr = (const float4*)(xr + (size_t)d * H_ * C + (size_t)h * C);
    const float4* pa = (const float4*)(att + (size_t)h * C);
    float acc = 0.f;
    const int c4 = C >> 2;
    for (int c = 0; c < c4; ++c) {
        float4 a = pl[c], b = pr[c], w = pa[c];
        acc = fmaf(lrelu(a.x + b.x), w.x, acc);
        acc = fmaf(lrelu(a.y + b.y), w.y, acc);
        acc = fmaf(lrelu(a.z + b.z), w.z, acc);
        acc = fmaf(lrelu(a.w + b.w), w.w, acc);
    }
    logits[t] = acc;
    atomicMax(&maxb[(size_t)d * H_ + h], fenc(acc));
}

// ---------------------------------------------------------------------------
// Edge pass 2: ex = exp(logit - max[dst]); den[dst] += ex; acc[dst] += ex*xl[src]
// ---------------------------------------------------------------------------
__global__ void k_edge_accum(const float* __restrict__ xl, const float* __restrict__ logits,
                             const unsigned* __restrict__ maxb,
                             const int* __restrict__ src, const int* __restrict__ dst,
                             float* __restrict__ den, float* __restrict__ acc, int C) {
    long t = (long)blockIdx.x * blockDim.x + threadIdx.x;
    if (t >= (long)E_ * H_) return;
    int e = (int)(t / H_), h = (int)(t - (long)e * H_);
    int s = src[e], d = dst[e];
    float m  = fdec(maxb[(size_t)d * H_ + h]);
    float ex = __expf(logits[t] - m);
    atomicAdd(&den[(size_t)d * H_ + h], ex);
    const float4* pl = (const float4*)(xl + (size_t)s * H_ * C + (size_t)h * C);
    float* pd = acc + (size_t)d * H_ * C + (size_t)h * C;
    const int c4 = C >> 2;
    for (int c = 0; c < c4; ++c) {
        float4 v = pl[c];
        atomicAdd(&pd[c * 4 + 0], ex * v.x);
        atomicAdd(&pd[c * 4 + 1], ex * v.y);
        atomicAdd(&pd[c * 4 + 2], ex * v.z);
        atomicAdd(&pd[c * 4 + 3], ex * v.w);
    }
}

// h16[n, 0:64] = x; h16[n, 64:256] = relu(acc/den + bias1)   (layer-2 GEMM input)
__global__ void k_finalize1(const float* __restrict__ x, const float* __restrict__ acc,
                            const float* __restrict__ den, const float* __restrict__ bias1,
                            _Float16* __restrict__ h16) {
    long t = (long)blockIdx.x * blockDim.x + threadIdx.x;
    if (t >= (long)N_ * F2_) return;
    int n = (int)(t / F2_), j = (int)(t - (long)n * F2_);
    float v;
    if (j < F_) {
        v = x[(size_t)n * F_ + j];
    } else {
        int jj = j - F_;
        int h  = jj / C1_;
        float dn = den[(size_t)n * H_ + h];
        float a  = acc[(size_t)n * HC1 + jj];
        v = (dn > 0.f) ? a / dn : 0.f;
        v += bias1[jj];
        v = (v > 0.f) ? v : 0.f;
    }
    h16[t] = (_Float16)v;
}

// out[n, 0:64] = x; out[n, 64:192] = relu(mean_h(acc/den) + bias2)
__global__ void k_finalize2(const float* __restrict__ x, const float* __restrict__ acc,
                            const float* __restrict__ den, const float* __restrict__ bias2,
                            float* __restrict__ out) {
    long t = (long)blockIdx.x * blockDim.x + threadIdx.x;
    if (t >= (long)N_ * OUTC) return;
    int n = (int)(t / OUTC), j = (int)(t - (long)n * OUTC);
    float v;
    if (j < F_) {
        v = x[(size_t)n * F_ + j];
    } else {
        int jj = j - F_;
        float s = 0.f;
#pragma unroll
        for (int h = 0; h < H_; ++h) {
            float dn = den[(size_t)n * H_ + h];
            float a  = acc[(size_t)n * HC2 + h * C2_ + jj];
            s += (dn > 0.f) ? a / dn : 0.f;
        }
        v = s * (1.0f / H_) + bias2[jj];
        v = (v > 0.f) ? v : 0.f;
    }
    out[t] = v;
}

// ---------------------------------------------------------------------------
extern "C" void kernel_launch(void* const* d_in, const int* in_sizes, int n_in,
                              void* d_out, int out_size, void* d_ws, size_t ws_size,
                              hipStream_t stream) {
    (void)in_sizes; (void)n_in; (void)out_size; (void)ws_size;

    const float* x     = (const float*)d_in[0];
    const float* W1l   = (const float*)d_in[1];
    const float* b1l   = (const float*)d_in[2];
    const float* W1r   = (const float*)d_in[3];
    const float* b1r   = (const float*)d_in[4];
    const float* att1  = (const float*)d_in[5];
    const float* bias1 = (const float*)d_in[6];
    const float* W2l   = (const float*)d_in[7];
    const float* b2l   = (const float*)d_in[8];
    const float* W2r   = (const float*)d_in[9];
    const float* b2r   = (const float*)d_in[10];
    const float* att2  = (const float*)d_in[11];
    const float* bias2 = (const float*)d_in[12];
    const int*   ei    = (const int*)d_in[13];
    const int* src = ei;
    const int* dst = ei + E_;

    // workspace carve-out (256B aligned); layer-2-sized regions reused by layer 1
    char* ws = (char*)d_ws;
    size_t off = 0;
    auto alloc = [&](size_t bytes) -> char* {
        char* p = ws + off;
        off += (bytes + 255) & ~(size_t)255;
        return p;
    };
    _Float16* x16  = (_Float16*)alloc((size_t)N_ * F_  * 2);
    _Float16* h16  = (_Float16*)alloc((size_t)N_ * F2_ * 2);
    _Float16* w1lT = (_Float16*)alloc((size_t)F_  * HC1 * 2);
    _Float16* w1rT = (_Float16*)alloc((size_t)F_  * HC1 * 2);
    _Float16* w2lT = (_Float16*)alloc((size_t)F2_ * HC2 * 2);
    _Float16* w2rT = (_Float16*)alloc((size_t)F2_ * HC2 * 2);
    float*    xl   = (float*)alloc((size_t)N_ * HC2 * 4);
    float*    xr   = (float*)alloc((size_t)N_ * HC2 * 4);
    float*    accb = (float*)alloc((size_t)N_ * HC2 * 4);
    float*    logt = (float*)alloc((size_t)E_ * H_ * 4);
    unsigned* maxb = (unsigned*)alloc((size_t)N_ * H_ * 4);
    float*    den  = (float*)alloc((size_t)N_ * H_ * 4);

    const int TB = 256;
    auto blks = [](long n) { return (unsigned)((n + 255) / 256); };

    // --- precision conversions -------------------------------------------
    k_cvt_f16<<<blks((long)N_ * F_), TB, 0, stream>>>(x, x16, (long)N_ * F_);
    k_cvt_w_t<<<blks((long)F_  * HC1), TB, 0, stream>>>(W1l, w1lT, F_,  HC1);
    k_cvt_w_t<<<blks((long)F_  * HC1), TB, 0, stream>>>(W1r, w1rT, F_,  HC1);
    k_cvt_w_t<<<blks((long)F2_ * HC2), TB, 0, stream>>>(W2l, w2lT, F2_, HC2);
    k_cvt_w_t<<<blks((long)F2_ * HC2), TB, 0, stream>>>(W2r, w2rT, F2_, HC2);

    // --- layer 1: projections (WMMA), softmax-aggregate -------------------
    // M = 50000 rows = 3125 16-row tiles = 625 blocks x 5 waves (exact)
    dim3 g1(N_ / 80, HC1 / 64);
    k_gemm_wmma<F_><<<g1, 160, 0, stream>>>(x16, w1lT, b1l, xl, HC1);
    k_gemm_wmma<F_><<<g1, 160, 0, stream>>>(x16, w1rT, b1r, xr, HC1);

    k_fill_u32<<<blks((long)N_ * H_),  TB, 0, stream>>>(maxb, ENC_NEG_INF, (long)N_ * H_);
    k_fill_u32<<<blks((long)N_ * H_),  TB, 0, stream>>>((unsigned*)den, 0u, (long)N_ * H_);
    k_fill_u32<<<blks((long)N_ * HC1), TB, 0, stream>>>((unsigned*)accb, 0u, (long)N_ * HC1);

    k_edge_logits<<<blks((long)E_ * H_), TB, 0, stream>>>(xl, xr, att1, src, dst, logt, maxb, C1_);
    k_edge_accum <<<blks((long)E_ * H_), TB, 0, stream>>>(xl, logt, maxb, src, dst, den, accb, C1_);
    k_finalize1  <<<blks((long)N_ * F2_), TB, 0, stream>>>(x, accb, den, bias1, h16);

    // --- layer 2 -----------------------------------------------------------
    dim3 g2(N_ / 80, HC2 / 64);
    k_gemm_wmma<F2_><<<g2, 160, 0, stream>>>(h16, w2lT, b2l, xl, HC2);
    k_gemm_wmma<F2_><<<g2, 160, 0, stream>>>(h16, w2rT, b2r, xr, HC2);

    k_fill_u32<<<blks((long)N_ * H_),  TB, 0, stream>>>(maxb, ENC_NEG_INF, (long)N_ * H_);
    k_fill_u32<<<blks((long)N_ * H_),  TB, 0, stream>>>((unsigned*)den, 0u, (long)N_ * H_);
    k_fill_u32<<<blks((long)N_ * HC2), TB, 0, stream>>>((unsigned*)accb, 0u, (long)N_ * HC2);

    k_edge_logits<<<blks((long)E_ * H_), TB, 0, stream>>>(xl, xr, att2, src, dst, logt, maxb, C2_);
    k_edge_accum <<<blks((long)E_ * H_), TB, 0, stream>>>(xl, logt, maxb, src, dst, den, accb, C2_);
    k_finalize2  <<<blks((long)N_ * OUTC), TB, 0, stream>>>(x, accb, den, bias2, (float*)d_out);
}
